// TransformerBlock_86681029968611
// MI455X (gfx1250) — compile-verified, matrix operand order checked by reading, not run on hardware
//
#include <hip/hip_runtime.h>
#include <hip/hip_bf16.h>
#include <math.h>

// ---------------------------------------------------------------------------
// Types for CDNA5 WMMA (wave32): v_wmma_f32_16x16x32_bf16
// ---------------------------------------------------------------------------
typedef __attribute__((ext_vector_type(16))) __bf16 bf16x16;
typedef __attribute__((ext_vector_type(8)))  float  f32x8;

union BF16Frag {
    bf16x16 v;
    uint4   q[2];  // two 16-byte halves (K 0..15 chunk, K 16..31 chunk)
};

__device__ __forceinline__ f32x8 wmma_bf16(const BF16Frag& a, const BF16Frag& b, f32x8 c) {
    // D = A(16x32 bf16) * B(32x16 bf16) + C(16x16 f32)
    return __builtin_amdgcn_wmma_f32_16x16x32_bf16(false, a.v, false, b.v,
                                                   (short)0, c, false, false);
}

__device__ __forceinline__ f32x8 zero8() {
    f32x8 z;
#pragma unroll
    for (int i = 0; i < 8; ++i) z[i] = 0.0f;
    return z;
}

__device__ __forceinline__ unsigned short f2bf(float f) {
    union { float f; unsigned int u; } c; c.f = f;
    unsigned int r = c.u + 0x7FFFu + ((c.u >> 16) & 1u);  // round-to-nearest-even
    return (unsigned short)(r >> 16);
}

// ---------------------------------------------------------------------------
// CDNA5 async global->LDS copy (GLOBAL_LOAD_ASYNC_TO_LDS_B128, ASYNCcnt).
// Each active lane copies 16 bytes: dsaddr from the lane's VDST VGPR,
// global addr from the VADDR pair (GV mode). Generic LDS pointers on amdgcn
// carry the wave-relative LDS offset in their low 32 bits.
// ---------------------------------------------------------------------------
__device__ __forceinline__ void async_load_b128(const void* g, void* lds) {
    unsigned ldsoff = (unsigned)(size_t)lds;
    unsigned long long ga = (unsigned long long)(size_t)g;
    asm volatile("global_load_async_to_lds_b128 %0, %1, off"
                 :: "v"(ldsoff), "v"(ga) : "memory");
}
__device__ __forceinline__ void wait_async0() {
    asm volatile("s_wait_asynccnt 0x0" ::: "memory");
}

// ---------------------------------------------------------------------------
// Problem constants
// ---------------------------------------------------------------------------
#define BB   4
#define SS   2048
#define DD   768
#define HH   12
#define FF   3072
#define DKH  64
#define MTOT (BB * SS)   // 8192 rows

// ---------------------------------------------------------------------------
// fp32 -> bf16 conversion
// ---------------------------------------------------------------------------
__global__ __launch_bounds__(256) void f32_to_bf16_kernel(const float* __restrict__ in,
                                                          unsigned short* __restrict__ out,
                                                          int n) {
    int i = blockIdx.x * 256 + threadIdx.x;
    if (i < n) out[i] = f2bf(in[i]);
}

// ---------------------------------------------------------------------------
// LayerNorm over last dim (D) -> bf16 output
// ---------------------------------------------------------------------------
__global__ __launch_bounds__(256) void ln_bf16_kernel(const float* __restrict__ x,
                                                      const float* __restrict__ g,
                                                      const float* __restrict__ b,
                                                      unsigned short* __restrict__ out,
                                                      int D) {
    __shared__ float red[256];
    const int row = blockIdx.x;
    const float* xr = x + (size_t)row * D;

    float s = 0.0f;
    for (int i = threadIdx.x; i < D; i += 256) s += xr[i];
    red[threadIdx.x] = s; __syncthreads();
    for (int o = 128; o > 0; o >>= 1) {
        if (threadIdx.x < o) red[threadIdx.x] += red[threadIdx.x + o];
        __syncthreads();
    }
    const float mean = red[0] / (float)D;
    __syncthreads();

    float vs = 0.0f;
    for (int i = threadIdx.x; i < D; i += 256) { float d = xr[i] - mean; vs += d * d; }
    red[threadIdx.x] = vs; __syncthreads();
    for (int o = 128; o > 0; o >>= 1) {
        if (threadIdx.x < o) red[threadIdx.x] += red[threadIdx.x + o];
        __syncthreads();
    }
    const float rstd = rsqrtf(red[0] / (float)D + 1e-5f);

    for (int i = threadIdx.x; i < D; i += 256)
        out[(size_t)row * D + i] = f2bf((xr[i] - mean) * rstd * g[i] + b[i]);
}

// ---------------------------------------------------------------------------
// Generic WMMA GEMM: C[M,N] = act(A[M,K] @ B[K,N] + bias) + residual
//   A, B bf16 row-major; bias fp32 [N] or null; residual fp32 [M,N] or null;
//   outF fp32 and/or outBF bf16. Block tile 128x128, K-step 64, 8 waves.
//   A tile streamed with async global->LDS; B tile transposed through VGPRs.
// ---------------------------------------------------------------------------
#define TSTR 72  // padded halves per LDS row (144B, 16B aligned)

__global__ __launch_bounds__(256) void gemm_bf16_kernel(
    const unsigned short* __restrict__ A, const unsigned short* __restrict__ Bw,
    const float* __restrict__ bias, const float* __restrict__ residual,
    float* __restrict__ outF, unsigned short* __restrict__ outBF,
    int M, int N, int K, int act) {

    __shared__ __align__(16) unsigned short As[128 * TSTR];  // A tile [128][64]
    __shared__ __align__(16) unsigned short Bs[128 * TSTR];  // B^T tile [n][64]

    const int tid   = threadIdx.x;
    const int wave  = tid >> 5;
    const int lane  = tid & 31;
    const int l16   = lane & 15;
    const int khalf = lane >> 4;
    const int wm    = wave >> 2;       // 0..1 -> M offset wm*64
    const int wn    = wave & 3;        // 0..3 -> N offset wn*32
    const int m0    = blockIdx.y * 128;
    const int n0    = blockIdx.x * 128;

    f32x8 acc[4][2];
#pragma unroll
    for (int i = 0; i < 4; ++i)
#pragma unroll
        for (int j = 0; j < 2; ++j) acc[i][j] = zero8();

    for (int k0 = 0; k0 < K; k0 += 64) {
        __syncthreads();
        // ---- A tile 128x64: 1024 chunks of 8 halves, async direct to LDS ----
#pragma unroll
        for (int j = 0; j < 4; ++j) {
            int c  = tid + j * 256;
            int r  = c >> 3, cg = c & 7;
            const unsigned short* src = A + (size_t)(m0 + r) * K + k0 + cg * 8;
            async_load_b128(src, As + r * TSTR + cg * 8);
            __builtin_prefetch(src + 64, 0, 1);  // speculative: next K tile
        }
        // ---- B tile 64x128, transposed into Bs[n][k] ----
#pragma unroll
        for (int j = 0; j < 4; ++j) {
            int c  = tid + j * 256;
            int kr = c >> 4, ng = c & 15;
            const unsigned short* src = Bw + (size_t)(k0 + kr) * N + n0 + ng * 8;
            uint4 d = *(const uint4*)src;
            __builtin_prefetch(src + (size_t)64 * N, 0, 1);
            const unsigned short* ds = (const unsigned short*)&d;
#pragma unroll
            for (int e = 0; e < 8; ++e) Bs[(ng * 8 + e) * TSTR + kr] = ds[e];
        }
        wait_async0();       // this wave's async copies landed in LDS
        __syncthreads();     // all waves' tiles visible

        // ---- 2 K-chunks of 32 -> 16 WMMAs per stage ----
#pragma unroll
        for (int ks = 0; ks < 2; ++ks) {
            BF16Frag a[4], b[2];
#pragma unroll
            for (int ti = 0; ti < 4; ++ti) {
                int row = wm * 64 + ti * 16 + l16;
                a[ti].q[0] = *(const uint4*)(As + row * TSTR + ks * 32 + khalf * 8);
                a[ti].q[1] = *(const uint4*)(As + row * TSTR + ks * 32 + 16 + khalf * 8);
            }
#pragma unroll
            for (int bj = 0; bj < 2; ++bj) {
                int col = wn * 32 + bj * 16 + l16;
                b[bj].q[0] = *(const uint4*)(Bs + col * TSTR + ks * 32 + khalf * 8);
                b[bj].q[1] = *(const uint4*)(Bs + col * TSTR + ks * 32 + 16 + khalf * 8);
            }
#pragma unroll
            for (int ti = 0; ti < 4; ++ti)
#pragma unroll
                for (int bj = 0; bj < 2; ++bj)
                    acc[ti][bj] = wmma_bf16(a[ti], b[bj], acc[ti][bj]);
        }
    }

    // ---- epilogue: C layout lane=col, VGPR r -> row r (+8 for high half) ----
#pragma unroll
    for (int bj = 0; bj < 2; ++bj) {
        int col = n0 + wn * 32 + bj * 16 + l16;
        float bsv = bias ? bias[col] : 0.0f;
#pragma unroll
        for (int ti = 0; ti < 4; ++ti) {
            int rb = m0 + wm * 64 + ti * 16 + khalf * 8;
#pragma unroll
            for (int r = 0; r < 8; ++r) {
                int row = rb + r;
                float v = acc[ti][bj][r] + bsv;
                if (act) v = 0.5f * v * (1.0f + erff(v * 0.70710678118654752f));
                if (residual) v += residual[(size_t)row * N + col];
                if (outF)  outF[(size_t)row * N + col]  = v;
                if (outBF) outBF[(size_t)row * N + col] = f2bf(v);
            }
        }
    }
}

// ---------------------------------------------------------------------------
// Flash attention (online softmax) with WMMA.
// Block: one (b,h), 128 query rows. 8 waves x 16 rows each. Key tiles of 64.
// q,k,v,ctx are bf16 in (B,S,D) layout with head slice [h*64 .. h*64+63].
// Q and K tiles use async global->LDS; V is transposed through VGPRs.
// ---------------------------------------------------------------------------
__global__ __launch_bounds__(256) void flash_attn_bf16_kernel(
    const unsigned short* __restrict__ q, const unsigned short* __restrict__ k,
    const unsigned short* __restrict__ v, const int* __restrict__ amask,
    unsigned short* __restrict__ ctx) {

    __shared__ __align__(16) unsigned short Qs[128 * TSTR];    // Q tile [128][64]
    __shared__ __align__(16) unsigned short Ks[64 * TSTR];     // K tile [t][dk]  (== Bt for q.k^T)
    __shared__ __align__(16) unsigned short Vs[DKH * TSTR];    // V^T   [dk][t]   (== Bt for p.v)
    __shared__ __align__(16) unsigned short Ps[8 * 16 * TSTR]; // per-wave P [16][64]

    const int tid   = threadIdx.x;
    const int wave  = tid >> 5;
    const int lane  = tid & 31;
    const int l16   = lane & 15;
    const int khalf = lane >> 4;
    const int b     = blockIdx.y / HH;
    const int h     = blockIdx.y % HH;
    const int q0    = blockIdx.x * 128;
    const size_t base = ((size_t)b * SS) * DD + h * DKH;

    // ---- load Q tile (128x64 halves = 1024 chunks of 8), async to LDS ----
#pragma unroll
    for (int j = 0; j < 4; ++j) {
        int c = tid + j * 256;
        int r = c >> 3, cg = c & 7;
        async_load_b128(q + base + (size_t)(q0 + r) * DD + cg * 8,
                        Qs + r * TSTR + cg * 8);
    }
    wait_async0();
    __syncthreads();

    // per-wave constant Q fragments (rows wave*16..+15), dk split into 2 K=32 chunks
    BF16Frag qa[2];
    {
        int row = wave * 16 + l16;
        qa[0].q[0] = *(const uint4*)(Qs + row * TSTR + khalf * 8);
        qa[0].q[1] = *(const uint4*)(Qs + row * TSTR + 16 + khalf * 8);
        qa[1].q[0] = *(const uint4*)(Qs + row * TSTR + 32 + khalf * 8);
        qa[1].q[1] = *(const uint4*)(Qs + row * TSTR + 48 + khalf * 8);
    }

    f32x8 O[4];
    float mrun[8], lrun[8];
#pragma unroll
    for (int i = 0; i < 4; ++i) O[i] = zero8();
#pragma unroll
    for (int r = 0; r < 8; ++r) { mrun[r] = -3.0e38f; lrun[r] = 0.0f; }

    const float sscale = 0.125f;  // 1/sqrt(64)

    for (int t0 = 0; t0 < SS; t0 += 64) {
        __syncthreads();
        // ---- K tile rows [t0..t0+63] async; V transposed via VGPRs ----
#pragma unroll
        for (int j = 0; j < 2; ++j) {
            int c = tid + j * 256;            // 512 chunks
            int r = c >> 3, cg = c & 7;
            async_load_b128(k + base + (size_t)(t0 + r) * DD + cg * 8,
                            Ks + r * TSTR + cg * 8);
            uint4 dv = *(const uint4*)(v + base + (size_t)(t0 + r) * DD + cg * 8);
            const unsigned short* ds = (const unsigned short*)&dv;
#pragma unroll
            for (int e = 0; e < 8; ++e) Vs[(cg * 8 + e) * TSTR + r] = ds[e];
        }
        wait_async0();
        __syncthreads();

        // ---- scores S = Q * K^T : 4 key n-tiles, dk = 2 K-steps ----
        f32x8 sacc[4];
#pragma unroll
        for (int nt = 0; nt < 4; ++nt) sacc[nt] = zero8();
#pragma unroll
        for (int ks = 0; ks < 2; ++ks) {
            BF16Frag bf[4];
#pragma unroll
            for (int nt = 0; nt < 4; ++nt) {
                int key = nt * 16 + l16;
                bf[nt].q[0] = *(const uint4*)(Ks + key * TSTR + ks * 32 + khalf * 8);
                bf[nt].q[1] = *(const uint4*)(Ks + key * TSTR + ks * 32 + 16 + khalf * 8);
            }
#pragma unroll
            for (int nt = 0; nt < 4; ++nt) sacc[nt] = wmma_bf16(qa[ks], bf[nt], sacc[nt]);
        }

        float maskadd[4];
#pragma unroll
        for (int nt = 0; nt < 4; ++nt) {
            int key = t0 + nt * 16 + l16;
            maskadd[nt] = amask[(size_t)b * SS + key] ? 0.0f : -1.0e30f;
        }

        // ---- online softmax per row (half-wave: rows r / r+8) ----
#pragma unroll
        for (int r = 0; r < 8; ++r) {
            float s0 = sacc[0][r] * sscale + maskadd[0];
            float s1 = sacc[1][r] * sscale + maskadd[1];
            float s2 = sacc[2][r] * sscale + maskadd[2];
            float s3 = sacc[3][r] * sscale + maskadd[3];
            float tm = fmaxf(fmaxf(s0, s1), fmaxf(s2, s3));
#pragma unroll
            for (int o = 1; o < 16; o <<= 1) tm = fmaxf(tm, __shfl_xor(tm, o, 16));
            float nm    = fmaxf(mrun[r], tm);
            float alpha = __expf(mrun[r] - nm);
            float p0 = __expf(s0 - nm), p1 = __expf(s1 - nm);
            float p2 = __expf(s2 - nm), p3 = __expf(s3 - nm);
            float rs = p0 + p1 + p2 + p3;
#pragma unroll
            for (int o = 1; o < 16; o <<= 1) rs += __shfl_xor(rs, o, 16);
            lrun[r] = lrun[r] * alpha + rs;
            mrun[r] = nm;
#pragma unroll
            for (int dt = 0; dt < 4; ++dt) O[dt][r] = O[dt][r] * alpha;
            // spill P (C layout) to wave-private LDS for A-fragment reload
            unsigned short* pr = Ps + (wave * 16 + r + khalf * 8) * TSTR;
            pr[0 * 16 + l16] = f2bf(p0);
            pr[1 * 16 + l16] = f2bf(p1);
            pr[2 * 16 + l16] = f2bf(p2);
            pr[3 * 16 + l16] = f2bf(p3);
        }

        // ---- ctx += P(16x64) @ V(64x64) ----
#pragma unroll
        for (int ks = 0; ks < 2; ++ks) {
            BF16Frag pa;
            int row = wave * 16 + l16;
            pa.q[0] = *(const uint4*)(Ps + row * TSTR + ks * 32 + khalf * 8);
            pa.q[1] = *(const uint4*)(Ps + row * TSTR + ks * 32 + 16 + khalf * 8);
#pragma unroll
            for (int dt = 0; dt < 4; ++dt) {
                BF16Frag vb;
                int d = dt * 16 + l16;
                vb.q[0] = *(const uint4*)(Vs + d * TSTR + ks * 32 + khalf * 8);
                vb.q[1] = *(const uint4*)(Vs + d * TSTR + ks * 32 + 16 + khalf * 8);
                O[dt] = wmma_bf16(pa, vb, O[dt]);
            }
        }
    }

    // ---- normalize and store ctx (B,S,D) bf16 ----
#pragma unroll
    for (int r = 0; r < 8; ++r) {
        float inv = 1.0f / lrun[r];
        int srow = q0 + wave * 16 + r + khalf * 8;
        size_t orow = ((size_t)b * SS + srow) * DD + h * DKH;
#pragma unroll
        for (int dt = 0; dt < 4; ++dt)
            ctx[orow + dt * 16 + l16] = f2bf(O[dt][r] * inv);
    }
}

// ---------------------------------------------------------------------------
// Host-side orchestration
// ---------------------------------------------------------------------------
extern "C" void kernel_launch(void* const* d_in, const int* in_sizes, int n_in,
                              void* d_out, int out_size, void* d_ws, size_t ws_size,
                              hipStream_t stream) {
    (void)in_sizes; (void)n_in; (void)out_size; (void)ws_size;
    const float* x     = (const float*)d_in[0];
    const int*   amask = (const int*)d_in[1];
    const float* ln1_g = (const float*)d_in[2];
    const float* ln1_b = (const float*)d_in[3];
    const float* Wq    = (const float*)d_in[4];
    const float* bq    = (const float*)d_in[5];
    const float* Wk    = (const float*)d_in[6];
    const float* bk    = (const float*)d_in[7];
    const float* Wv    = (const float*)d_in[8];
    const float* bv    = (const float*)d_in[9];
    const float* Wo    = (const float*)d_in[10];
    const float* bo    = (const float*)d_in[11];
    const float* ln2_g = (const float*)d_in[12];
    const float* ln2_b = (const float*)d_in[13];
    const float* W1    = (const float*)d_in[14];
    const float* b1    = (const float*)d_in[15];
    const float* W2    = (const float*)d_in[16];
    const float* b2    = (const float*)d_in[17];
    float* out = (float*)d_out;

    unsigned char* ws = (unsigned char*)d_ws;
    const size_t HB = (size_t)MTOT * DD * 2;           // bf16 (M,D) buffer bytes
    unsigned short* h1   = (unsigned short*)(ws + 0 * HB);
    unsigned short* qb   = (unsigned short*)(ws + 1 * HB);
    unsigned short* kb   = (unsigned short*)(ws + 2 * HB);
    unsigned short* vb   = (unsigned short*)(ws + 3 * HB);
    unsigned short* ctx  = (unsigned short*)(ws + 4 * HB);
    unsigned short* h2   = (unsigned short*)(ws + 5 * HB);
    unsigned short* act  = (unsigned short*)(ws + 6 * HB);  // (M,F) bf16 = 4*HB
    float*          xatt = (float*)(ws + 10 * HB);          // (M,D) fp32 = 2*HB
    unsigned short* wqb  = (unsigned short*)(ws + 12 * HB);
    unsigned short* wkb  = wqb + (size_t)DD * DD;
    unsigned short* wvb  = wkb + (size_t)DD * DD;
    unsigned short* wob  = wvb + (size_t)DD * DD;
    unsigned short* w1b  = wob + (size_t)DD * DD;
    unsigned short* w2b  = w1b + (size_t)DD * FF;

    const int nDD = DD * DD, nDF = DD * FF;
    f32_to_bf16_kernel<<<(nDD + 255) / 256, 256, 0, stream>>>(Wq, wqb, nDD);
    f32_to_bf16_kernel<<<(nDD + 255) / 256, 256, 0, stream>>>(Wk, wkb, nDD);
    f32_to_bf16_kernel<<<(nDD + 255) / 256, 256, 0, stream>>>(Wv, wvb, nDD);
    f32_to_bf16_kernel<<<(nDD + 255) / 256, 256, 0, stream>>>(Wo, wob, nDD);
    f32_to_bf16_kernel<<<(nDF + 255) / 256, 256, 0, stream>>>(W1, w1b, nDF);
    f32_to_bf16_kernel<<<(nDF + 255) / 256, 256, 0, stream>>>(W2, w2b, nDF);

    // LN1 -> h1 (bf16)
    ln_bf16_kernel<<<MTOT, 256, 0, stream>>>(x, ln1_g, ln1_b, h1, DD);

    // QKV projections: (8192,768) = (8192,768)@(768,768) + b
    dim3 gD(DD / 128, MTOT / 128);  // (6, 64)
    gemm_bf16_kernel<<<gD, 256, 0, stream>>>(h1, wqb, bq, nullptr, nullptr, qb, MTOT, DD, DD, 0);
    gemm_bf16_kernel<<<gD, 256, 0, stream>>>(h1, wkb, bk, nullptr, nullptr, kb, MTOT, DD, DD, 0);
    gemm_bf16_kernel<<<gD, 256, 0, stream>>>(h1, wvb, bv, nullptr, nullptr, vb, MTOT, DD, DD, 0);

    // Flash attention -> ctx (bf16, (B,S,D))
    flash_attn_bf16_kernel<<<dim3(SS / 128, BB * HH), 256, 0, stream>>>(qb, kb, vb, amask, ctx);

    // Output projection + residual: xattn = x + ctx@Wo + bo  (fp32)
    gemm_bf16_kernel<<<gD, 256, 0, stream>>>(ctx, wob, bo, x, xatt, nullptr, MTOT, DD, DD, 0);

    // LN2 -> h2 (bf16)
    ln_bf16_kernel<<<MTOT, 256, 0, stream>>>(xatt, ln2_g, ln2_b, h2, DD);

    // MLP: act = gelu(h2@W1 + b1)  (bf16), out = xattn + act@W2 + b2 (fp32)
    dim3 gF(FF / 128, MTOT / 128);  // (24, 64)
    gemm_bf16_kernel<<<gF, 256, 0, stream>>>(h2, w1b, b1, nullptr, nullptr, act, MTOT, FF, DD, 1);
    gemm_bf16_kernel<<<gD, 256, 0, stream>>>(act, w2b, b2, xatt, out, nullptr, MTOT, DD, FF, 0);
}
